// ToxGNN_82927228551354
// MI455X (gfx1250) — compile-verified, matrix operand order checked by reading, not run on hardware
//
#include <hip/hip_runtime.h>

// ---------------------------------------------------------------------------
// ToxGNN (3x GINConv + mean-pool + head) for MI455X / gfx1250.
// bf16 activations after layer-1's first matmul:
//   - halves HBM traffic in the memory-bound edge gather/scatter phase
//   - feeds v_wmma_f32_16x16x32_bf16 directly (fp32 accumulate)
// Edge scatter uses global_atomic_pk_add_bf16 (CDNA5 packed bf16 atomic).
// GEMM: 128x128 block tile, K-step 32, software-pipelined global->LDS->WMMA.
// Prefetch staging uses named scalars (not arrays/lambda) so SROA keeps them
// in VGPRs -- avoids the promote-alloca-to-LDS round trip seen in round 2.
// ---------------------------------------------------------------------------

typedef unsigned short u16;
typedef unsigned int   u32;

typedef __attribute__((ext_vector_type(16))) __bf16 v16bf;
typedef __attribute__((ext_vector_type(8)))  float  v8f;

#define NNODES 100000
#define NEDGES 1600000
#define NGRAPH 4096
#define HID    512

// ---- bf16 <-> f32 helpers (RNE) -------------------------------------------
static __device__ __forceinline__ u16 f2bf(float f) {
    union { float f; u32 u; } v; v.f = f;
    u32 r = v.u + 0x7FFFu + ((v.u >> 16) & 1u);
    return (u16)(r >> 16);
}
static __device__ __forceinline__ float bf2f(u16 h) {
    union { u32 u; float f; } v; v.u = ((u32)h) << 16;
    return v.f;
}

// ---- packed bf16 atomic add (CDNA5 GLOBAL_ATOMIC_PK_ADD_BF16) -------------
static __device__ __forceinline__ void atomic_pk_add_bf16(u32* addr, u32 data) {
    asm volatile("global_atomic_pk_add_bf16 %0, %1, off"
                 :: "v"(addr), "v"(data) : "memory");
}

// ===========================================================================
// small utility kernels
// ===========================================================================
__global__ void k_f32_to_bf16(const float* __restrict__ in, u16* __restrict__ out, int n) {
    int i = blockIdx.x * blockDim.x + threadIdx.x;
    if (i < n) out[i] = f2bf(in[i]);
}

__global__ void k_copy_f32(const float* __restrict__ in, float* __restrict__ out, int n) {
    int i = blockIdx.x * blockDim.x + threadIdx.x;
    if (i < n) out[i] = in[i];
}

__global__ void k_copy_u4(const uint4* __restrict__ in, uint4* __restrict__ out, int n) {
    int i = blockIdx.x * blockDim.x + threadIdx.x;
    if (i < n) out[i] = in[i];
}

__global__ void k_zero_f32(float* __restrict__ p, int n) {
    int i = blockIdx.x * blockDim.x + threadIdx.x;
    if (i < n) p[i] = 0.0f;
}

// ===========================================================================
// layer-1 aggregation (7-dim, fp32) + K=7 input GEMM (scalar; 0.7 GF)
// ===========================================================================
__global__ void k_agg7_edges(const float* __restrict__ x, float* __restrict__ agg,
                             const int* __restrict__ src, const int* __restrict__ dst) {
    int e = blockIdx.x * blockDim.x + threadIdx.x;
    if (e >= NEDGES) return;
    int s = src[e], d = dst[e];
#pragma unroll
    for (int k = 0; k < 7; ++k)
        atomicAdd(&agg[d * 7 + k], x[s * 7 + k]);
}

__global__ void k_lin7(const float* __restrict__ agg, const float* __restrict__ w,
                       const float* __restrict__ b, u16* __restrict__ out) {
    int idx = blockIdx.x * blockDim.x + threadIdx.x;
    if (idx >= NNODES * HID) return;
    int m = idx >> 9;
    int n = idx & (HID - 1);
    float s = b[n];
#pragma unroll
    for (int k = 0; k < 7; ++k)
        s += agg[m * 7 + k] * w[k * HID + n];
    out[idx] = f2bf(s > 0.f ? s : 0.f);
}

// ===========================================================================
// bf16 edge aggregation: one wave per edge, 512 features (8 dwords/lane)
// ===========================================================================
__global__ void k_agg_edges_bf16(const u16* __restrict__ h, u16* __restrict__ agg,
                                 const int* __restrict__ src, const int* __restrict__ dst) {
    int gid = blockIdx.x * blockDim.x + threadIdx.x;
    int e = gid >> 5;
    int lane = gid & 31;
    if (e >= NEDGES) return;
    int s = src[e], d = dst[e];
    const u32* hs = (const u32*)(h + (size_t)s * HID);
    u32* ad = (u32*)(agg + (size_t)d * HID);
#pragma unroll
    for (int i = 0; i < 8; ++i) {
        u32 data = hs[lane + i * 32];
        atomic_pk_add_bf16(&ad[lane + i * 32], data);
    }
}

// ===========================================================================
// WMMA bf16 GEMM: Y(MxN,bf16) = relu(X(MxK,bf16) @ W(KxN,bf16) + bias(f32))
// block tile 128x128, K-step 32, 8 waves x (2x4) 16x16 tiles, pipelined.
// Requires: N % 128 == 0, K % 32 == 0 (512/256 here); M arbitrary.
// ===========================================================================
#define LDSS 40   // padded LDS row stride (ushorts): 80B rows, 16B-aligned chunks

struct B32x2 { uint4 lo, hi; };

__global__ __launch_bounds__(256)
void k_gemm_bf16(const u16* __restrict__ X, const u16* __restrict__ W,
                 const float* __restrict__ bias, u16* __restrict__ Y,
                 int M, int N, int K) {
    __shared__ u16 As[128 * LDSS];
    __shared__ u16 Bs[128 * LDSS];

    const int tid  = threadIdx.x;
    const int lane = tid & 31;
    const int wave = tid >> 5;
    const int m0 = blockIdx.x * 128;
    const int n0 = blockIdx.y * 128;

    const int half = lane >> 4;   // K-half select (ISA 7.12.2 16-bit A layout)
    const int r16  = lane & 15;   // row within 16x16 tile

    const int wr = wave >> 1;     // 0..3 -> row-tile pair
    const int wc = wave & 1;      // 0..1 -> col-tile quad

    // ---- A-tile mapping: 128x32 bf16 = 512 uint4; 2 per thread ------------
    // lin = tid + i*256: r = lin>>2 (0..127), q = lin&3 (16B chunk in row)
    int aG0, aG1, aL0, aL1;
    {
        int r = tid >> 2, q = tid & 3;
        int m = m0 + r; if (m > M - 1) m = M - 1;   // clamp: rows >= M never stored
        aG0 = m * (K / 8) + q;                      // uint4 index; +k0/8 per step
        aL0 = r * LDSS + 8 * q;                     // ushort index
        r = (tid + 256) >> 2; q = tid & 3;          // (tid+256)&3 == tid&3
        m = m0 + r; if (m > M - 1) m = M - 1;
        aG1 = m * (K / 8) + q;
        aL1 = r * LDSS + 8 * q;
    }
    // ---- B-tile mapping: 32x128 bf16, 2x4 bf16 micro-transpose per item ---
    // lin = tid + i*256: kp = lin>>5 (k-pair 0..15), nq = lin&31 (n-quad)
    int bG0, bG1, bL0, bL1;
    {
        int kp = tid >> 5, nq = tid & 31;
        bG0 = (2 * kp) * (N / 4) + n0 / 4 + nq;     // uint2 index; +k0*(N/4)
        bL0 = (4 * nq) * LDSS + 2 * kp;             // ushort index (Bs[n][k])
        kp = (tid + 256) >> 5; nq = tid & 31;
        bG1 = (2 * kp) * (N / 4) + n0 / 4 + nq;
        bL1 = (4 * nq) * LDSS + 2 * kp;
    }

    const uint4* Xg4 = (const uint4*)X;
    const uint2* Wg2 = (const uint2*)W;

    // prefetch staging: named scalars only -> guaranteed VGPRs
    uint4 aR0, aR1;
    uint2 bA0, bA1, bB0, bB1;
#define GPREF(k0)                                       \
    do {                                                \
        aR0 = Xg4[aG0 + ((k0) >> 3)];                   \
        aR1 = Xg4[aG1 + ((k0) >> 3)];                   \
        bA0 = Wg2[bG0 + (k0) * (N / 4)];                \
        bA1 = Wg2[bG0 + (k0) * (N / 4) + (N / 4)];      \
        bB0 = Wg2[bG1 + (k0) * (N / 4)];                \
        bB1 = Wg2[bG1 + (k0) * (N / 4) + (N / 4)];      \
    } while (0)

    GPREF(0);

    v8f acc[2][4];
#pragma unroll
    for (int i = 0; i < 2; ++i)
#pragma unroll
        for (int j = 0; j < 4; ++j)
            acc[i][j] = (v8f){0.f,0.f,0.f,0.f,0.f,0.f,0.f,0.f};

    for (int k0 = 0; k0 < K; k0 += 32) {
        // commit prefetched slice to LDS
        *(uint4*)&As[aL0] = aR0;
        *(uint4*)&As[aL1] = aR1;
        {
            u32 d00 = bA0.x, d01 = bA0.y, d10 = bA1.x, d11 = bA1.y;
            *(u32*)&Bs[bL0]            = (d00 & 0xffffu) | (d10 << 16);
            *(u32*)&Bs[bL0 + LDSS]     = (d00 >> 16)     | (d10 & 0xffff0000u);
            *(u32*)&Bs[bL0 + 2*LDSS]   = (d01 & 0xffffu) | (d11 << 16);
            *(u32*)&Bs[bL0 + 3*LDSS]   = (d01 >> 16)     | (d11 & 0xffff0000u);
        }
        {
            u32 d00 = bB0.x, d01 = bB0.y, d10 = bB1.x, d11 = bB1.y;
            *(u32*)&Bs[bL1]            = (d00 & 0xffffu) | (d10 << 16);
            *(u32*)&Bs[bL1 + LDSS]     = (d00 >> 16)     | (d10 & 0xffff0000u);
            *(u32*)&Bs[bL1 + 2*LDSS]   = (d01 & 0xffffu) | (d11 << 16);
            *(u32*)&Bs[bL1 + 3*LDSS]   = (d01 >> 16)     | (d11 & 0xffff0000u);
        }
        __syncthreads();

        // overlap next slice's global loads with this slice's WMMAs
        if (k0 + 32 < K) { GPREF(k0 + 32); }

        v16bf afrag[2];
#pragma unroll
        for (int i = 0; i < 2; ++i) {
            int arow = (2 * wr + i) * 16 + r16;
            B32x2 t;
            t.lo = *(const uint4*)&As[arow * LDSS + half * 8];        // K 0-7 / 8-15
            t.hi = *(const uint4*)&As[arow * LDSS + 16 + half * 8];   // K 16-23 / 24-31
            afrag[i] = __builtin_bit_cast(v16bf, t);
        }
        v16bf bfrag[4];
#pragma unroll
        for (int j = 0; j < 4; ++j) {
            int bcol = (4 * wc + j) * 16 + r16;
            B32x2 t;
            t.lo = *(const uint4*)&Bs[bcol * LDSS + half * 8];
            t.hi = *(const uint4*)&Bs[bcol * LDSS + 16 + half * 8];
            bfrag[j] = __builtin_bit_cast(v16bf, t);
        }
#pragma unroll
        for (int i = 0; i < 2; ++i)
#pragma unroll
            for (int j = 0; j < 4; ++j)
                acc[i][j] = __builtin_amdgcn_wmma_f32_16x16x32_bf16(
                    false, afrag[i], false, bfrag[j],
                    (short)0, acc[i][j], false, false);
        __syncthreads();
    }
#undef GPREF

    // epilogue: C/D layout -> lane<16: rows 0-7, lane>=16: rows 8-15; col=lane&15
#pragma unroll
    for (int i = 0; i < 2; ++i) {
        int mt = m0 + (2 * wr + i) * 16 + half * 8;
#pragma unroll
        for (int j = 0; j < 4; ++j) {
            int n = n0 + (4 * wc + j) * 16 + r16;
            float bv = bias[n];
#pragma unroll
            for (int e = 0; e < 8; ++e) {
                int m = mt + e;
                if (m < M) {
                    float v = acc[i][j][e] + bv;
                    Y[(size_t)m * N + n] = f2bf(v > 0.f ? v : 0.f);
                }
            }
        }
    }
}

// ===========================================================================
// mean-pool + head
// ===========================================================================
__global__ void k_pool_sum(const u16* __restrict__ h, const int* __restrict__ batch,
                           float* __restrict__ sums, float* __restrict__ cnts) {
    int gid = blockIdx.x * blockDim.x + threadIdx.x;
    int node = gid >> 5;
    int lane = gid & 31;
    if (node >= NNODES) return;
    int g = batch[node];
    const u16* hp = h + (size_t)node * HID;
    float* sp = sums + (size_t)g * HID;
#pragma unroll
    for (int i = 0; i < 16; ++i)
        atomicAdd(&sp[lane + i * 32], bf2f(hp[lane + i * 32]));
    if (lane == 0) atomicAdd(&cnts[g], 1.0f);
}

__global__ void k_pool_div(const float* __restrict__ sums, const float* __restrict__ cnts,
                           u16* __restrict__ pooled) {
    int idx = blockIdx.x * blockDim.x + threadIdx.x;
    if (idx >= NGRAPH * HID) return;
    int g = idx >> 9;
    float c = cnts[g];
    c = c < 1.0f ? 1.0f : c;
    pooled[idx] = f2bf(sums[idx] / c);
}

__global__ void k_final(const u16* __restrict__ mid, const float* __restrict__ w,
                        const float* __restrict__ b, float* __restrict__ out) {
    int idx = blockIdx.x * blockDim.x + threadIdx.x;
    if (idx >= NGRAPH * 12) return;
    int g = idx / 12;
    int o = idx % 12;
    float s = b[o];
    const u16* mp = mid + (size_t)g * 256;
    for (int k = 0; k < 256; ++k)
        s += bf2f(mp[k]) * w[k * 12 + o];
    out[idx] = s;
}

// ===========================================================================
// launch
// ===========================================================================
static inline size_t alignup(size_t x) { return (x + 255) & ~(size_t)255; }

extern "C" void kernel_launch(void* const* d_in, const int* in_sizes, int n_in,
                              void* d_out, int out_size, void* d_ws, size_t ws_size,
                              hipStream_t stream) {
    const float* x     = (const float*)d_in[0];
    const int*   ei    = (const int*)d_in[1];
    const int*   src   = ei;
    const int*   dst   = ei + NEDGES;
    const int*   batch = (const int*)d_in[2];
    const float* c1_w1 = (const float*)d_in[3];  const float* c1_b1 = (const float*)d_in[4];
    const float* c1_w2 = (const float*)d_in[5];  const float* c1_b2 = (const float*)d_in[6];
    const float* c2_w1 = (const float*)d_in[7];  const float* c2_b1 = (const float*)d_in[8];
    const float* c2_w2 = (const float*)d_in[9];  const float* c2_b2 = (const float*)d_in[10];
    const float* c3_w1 = (const float*)d_in[11]; const float* c3_b1 = (const float*)d_in[12];
    const float* c3_w2 = (const float*)d_in[13]; const float* c3_b2 = (const float*)d_in[14];
    const float* lw1   = (const float*)d_in[15]; const float* lb1   = (const float*)d_in[16];
    const float* lw2   = (const float*)d_in[17]; const float* lb2   = (const float*)d_in[18];

    // workspace layout
    char* p = (char*)d_ws;
    const size_t NH   = (size_t)NNODES * HID;         // node-feature elements
    u16* hA = (u16*)p;           p += alignup(NH * 2);
    u16* hB = (u16*)p;           p += alignup(NH * 2);
    u16* hC = (u16*)p;           p += alignup(NH * 2);
    float* agg7 = (float*)p;     p += alignup((size_t)NNODES * 7 * 4);
    u16* w2_1b = (u16*)p;        p += alignup(512 * 512 * 2);
    u16* w1_2b = (u16*)p;        p += alignup(512 * 512 * 2);
    u16* w2_2b = (u16*)p;        p += alignup(512 * 512 * 2);
    u16* w1_3b = (u16*)p;        p += alignup(512 * 512 * 2);
    u16* w2_3b = (u16*)p;        p += alignup(512 * 512 * 2);
    u16* lw1b  = (u16*)p;        p += alignup(512 * 256 * 2);
    float* sums = (float*)p;     p += (size_t)NGRAPH * HID * 4;   // keep cnts adjacent
    float* cnts = (float*)p;     p += alignup((size_t)NGRAPH * 4);
    u16* mid   = (u16*)p;        p += alignup((size_t)NGRAPH * 256 * 2);

    const int B = 256;
    auto blk = [](long n, int b) { return (int)((n + b - 1) / b); };

    // ---- weight conversion (fp32 -> bf16), ~1.44M elements -----------------
    k_f32_to_bf16<<<blk(262144, B), B, 0, stream>>>(c1_w2, w2_1b, 262144);
    k_f32_to_bf16<<<blk(262144, B), B, 0, stream>>>(c2_w1, w1_2b, 262144);
    k_f32_to_bf16<<<blk(262144, B), B, 0, stream>>>(c2_w2, w2_2b, 262144);
    k_f32_to_bf16<<<blk(262144, B), B, 0, stream>>>(c3_w1, w1_3b, 262144);
    k_f32_to_bf16<<<blk(262144, B), B, 0, stream>>>(c3_w2, w2_3b, 262144);
    k_f32_to_bf16<<<blk(131072, B), B, 0, stream>>>(lw1,   lw1b,  131072);

    dim3 gN512(blk(NNODES, 128), HID / 128);  // 782 x 4
    dim3 gP256(blk(NGRAPH, 128), 256 / 128);  // 32 x 2

    // ---- layer 1 ----------------------------------------------------------
    k_copy_f32<<<blk(NNODES * 7, B), B, 0, stream>>>(x, agg7, NNODES * 7);
    k_agg7_edges<<<blk(NEDGES, B), B, 0, stream>>>(x, agg7, src, dst);
    k_lin7<<<blk((long)NNODES * HID, B), B, 0, stream>>>(agg7, c1_w1, c1_b1, hA);
    k_gemm_bf16<<<gN512, B, 0, stream>>>(hA, w2_1b, c1_b2, hB, NNODES, HID, HID);

    // ---- layer 2 ----------------------------------------------------------
    k_copy_u4<<<blk(NH / 8, B), B, 0, stream>>>((const uint4*)hB, (uint4*)hC, (int)(NH / 8));
    k_agg_edges_bf16<<<blk((long)NEDGES * 32, B), B, 0, stream>>>(hB, hC, src, dst);
    k_gemm_bf16<<<gN512, B, 0, stream>>>(hC, w1_2b, c2_b1, hA, NNODES, HID, HID);
    k_gemm_bf16<<<gN512, B, 0, stream>>>(hA, w2_2b, c2_b2, hB, NNODES, HID, HID);

    // ---- layer 3 ----------------------------------------------------------
    k_copy_u4<<<blk(NH / 8, B), B, 0, stream>>>((const uint4*)hB, (uint4*)hC, (int)(NH / 8));
    k_agg_edges_bf16<<<blk((long)NEDGES * 32, B), B, 0, stream>>>(hB, hC, src, dst);
    k_gemm_bf16<<<gN512, B, 0, stream>>>(hC, w1_3b, c3_b1, hA, NNODES, HID, HID);
    k_gemm_bf16<<<gN512, B, 0, stream>>>(hA, w2_3b, c3_b2, hB, NNODES, HID, HID);

    // ---- pool + head ------------------------------------------------------
    k_zero_f32<<<blk(NGRAPH * HID + NGRAPH, B), B, 0, stream>>>(sums, NGRAPH * HID + NGRAPH);
    k_pool_sum<<<blk((long)NNODES * 32, B), B, 0, stream>>>(hB, batch, sums, cnts);
    k_pool_div<<<blk(NGRAPH * HID, B), B, 0, stream>>>(sums, cnts, hC);
    k_gemm_bf16<<<gP256, B, 0, stream>>>(hC, lw1b, lb1, mid, NGRAPH, 256, HID);
    k_final<<<blk(NGRAPH * 12, B), B, 0, stream>>>(mid, lw2, lb2, (float*)d_out);
}
